// DenseKAN_15659450761917
// MI455X (gfx1250) — matrix operand order repaired and checked by reading, not compiled, not run on hardware
//
#include <hip/hip_runtime.h>
#include <hip/hip_bf16.h>

// ---------------------------------------------------------------------------
// DenseKAN as  out = A_aug @ W_fused + bias
//   A_aug [1024 x 2304]  bf16 :  A[b, k*256+i] = spline basis k of x[b,i]
//                                A[b, 8*256+i] = silu(x[b,i])
//   W_fused stored N-major as Wt[256][2304] bf16:
//                                Wt[o, k*256+i] = spline_kernel[i,k,o]*scale[i,o]
//                                Wt[o, 8*256+i] = scale[i,o]
// GEMM: v_wmma_f32_16x16x32_bf16 (wave32), double-buffered LDS fed by
// global_load_async_to_lds_b128 (ASYNCcnt), one barrier per K-step of 64.
// ---------------------------------------------------------------------------

#define BATCH     1024
#define IN_DIM    256
#define UNITS     256
#define KAUG      (IN_DIM * 9)      // 2304
#define KT        64                // K step per iteration (2 WMMA K-slices)
#define NIT       (KAUG / KT)       // 36
#define BM        128               // block tile M (4 waves)
#define BN        64                // block tile N (2 waves)

typedef __attribute__((ext_vector_type(16))) __bf16 v16bf;
typedef __attribute__((ext_vector_type(8)))  float  v8f;

union FragBf { uint4 u[2]; v16bf v; };

// ------------------------- kernel 1a: activations --------------------------
// Uniform grid: 12 knots, g[t] = -2.2 + 0.4 t (spacing h = 0.4); all
// Cox-de Boor denominators are exactly k*h.
__global__ void kan_prep_act(const float* __restrict__ x,
                             __hip_bfloat16* __restrict__ A) {
    int idx = blockIdx.x * 256 + threadIdx.x;      // one thread per (b,i)
    if (idx >= BATCH * IN_DIM) return;
    int b = idx >> 8;
    int i = idx & 255;
    float xv = x[(size_t)b * IN_DIM + i];

    const float g0 = -2.2f;
    const float h  = 0.4f;

    float bb[11];
#pragma unroll
    for (int j = 0; j < 11; ++j) {
        float gj  = g0 + h * (float)j;
        float gj1 = g0 + h * (float)(j + 1);
        bb[j] = (xv >= gj && xv < gj1) ? 1.0f : 0.0f;
    }
#pragma unroll
    for (int k = 1; k <= 3; ++k) {
        float inv = 1.0f / (h * (float)k);
#pragma unroll
        for (int j = 0; j + k < 11; ++j) {
            float left  = (xv - (g0 + h * (float)j)) * inv;
            float right = ((g0 + h * (float)(j + k + 1)) - xv) * inv;
            bb[j] = left * bb[j] + right * bb[j + 1];
        }
    }
    float si = xv / (1.0f + __expf(-xv));          // silu

    size_t base = (size_t)b * KAUG;
#pragma unroll
    for (int k = 0; k < 8; ++k)
        A[base + (size_t)k * IN_DIM + i] = __float2bfloat16(bb[k]);
    A[base + (size_t)8 * IN_DIM + i] = __float2bfloat16(si);
}

// ------------------------- kernel 1b: fused weights ------------------------
__global__ void kan_prep_w(const float* __restrict__ sk,      // [256][8][256]
                           const float* __restrict__ scale,   // [256][256]
                           __hip_bfloat16* __restrict__ Wt) {
    int idx = blockIdx.x * 256 + threadIdx.x;      // one thread per (o,kcomb)
    if (idx >= UNITS * KAUG) return;
    int o  = idx / KAUG;
    int kc = idx - o * KAUG;
    int k  = kc >> 8;
    int i  = kc & 255;
    float s = scale[(size_t)i * UNITS + o];
    float w = (k < 8) ? sk[(size_t)i * (8 * UNITS) + (size_t)k * UNITS + o] * s
                      : s;
    Wt[(size_t)o * KAUG + kc] = __float2bfloat16(w);
}

// ------------------------------ kernel 2: GEMM -----------------------------
// Async copy of one 16B chunk: global -> LDS, tracked by ASYNCcnt.
// Low 32 bits of a generic LDS pointer ARE the LDS byte address (ISA 10.2).
__device__ __forceinline__ void async_ld16(void* lds_ptr, const void* gptr) {
    unsigned lds_addr = (unsigned)(unsigned long long)lds_ptr;
    asm volatile("global_load_async_to_lds_b128 %0, %1, off"
                 :: "v"(lds_addr), "v"(gptr)
                 : "memory");
}
__device__ __forceinline__ void wait_async0() {
    asm volatile("s_wait_asynccnt 0x0" ::: "memory");
}

__device__ __forceinline__ void store_ctile(float* __restrict__ out,
                                            const float* __restrict__ bias,
                                            const v8f& c, int grow0, int gcol) {
    float bv = bias[gcol];
#pragma unroll
    for (int j = 0; j < 8; ++j)
        out[(size_t)(grow0 + j) * UNITS + gcol] = c[j] + bv;
}

__global__ __launch_bounds__(256)
void kan_gemm(const __hip_bfloat16* __restrict__ A,    // [1024][2304]
              const __hip_bfloat16* __restrict__ Wt,   // [256][2304]
              const float* __restrict__ bias,          // [256]
              float* __restrict__ out) {               // [1024][256]
    __shared__ __align__(16) unsigned short As[2][BM * KT];   // 2 x 16 KB
    __shared__ __align__(16) unsigned short Bs[2][BN * KT];   // 2 x  8 KB

    const int tid    = threadIdx.x;
    const int lane   = tid & 31;
    const int wave   = tid >> 5;          // 0..7
    const int wave_m = wave & 3;          // 4 waves along M
    const int wave_n = wave >> 2;         // 2 waves along N
    const int bm     = blockIdx.x * BM;   // grid.x = 8
    const int bn     = blockIdx.y * BN;   // grid.y = 4

    v8f c00 = {0.f,0.f,0.f,0.f,0.f,0.f,0.f,0.f};
    v8f c01 = c00, c10 = c00, c11 = c00;

    const int rbase = wave_m * 32;
    const int cbase = wave_n * 32;
    const int rlo   = lane & 15;          // row/col within 16-wide fragment
    const int sel   = lane >> 4;          // K-interleave select (wave32 layout)

    // Per-thread staging coordinates (8 x 16B chunks per 64-elem row/col).
    const int ar = tid >> 3;              // A rows 0..31 (+32/step), 4 steps
    const int aq = tid & 7;               // chunk-in-row
    const int bc = tid >> 3;              // B cols 0..31 (+32), 2 steps
    const int bq = tid & 7;

    // ---- stage one K-slab [KT wide] into buffer `buf` (async) ----
    auto stage = [&](int buf, int kbase) {
#pragma unroll
        for (int j = 0; j < 4; ++j) {                 // A: 128 rows
            int row = ar + 32 * j;
            async_ld16(&As[buf][row * KT + aq * 8],
                       A + ((size_t)(bm + row) * KAUG + kbase + aq * 8));
        }
#pragma unroll
        for (int j = 0; j < 2; ++j) {                 // B: 64 cols (N-major)
            int col = bc + 32 * j;
            async_ld16(&Bs[buf][col * KT + bq * 8],
                       Wt + ((size_t)(bn + col) * KAUG + kbase + bq * 8));
        }
    };

    stage(0, 0);
    wait_async0();
    __syncthreads();

    for (int it = 0; it < NIT; ++it) {
        const int buf = it & 1;
        if (it + 1 < NIT) stage(buf ^ 1, (it + 1) * KT);   // overlap with math

        const uint4* As4 = reinterpret_cast<const uint4*>(&As[buf][0]);
        const uint4* Bs4 = reinterpret_cast<const uint4*>(&Bs[buf][0]);

#pragma unroll
        for (int kh = 0; kh < KT; kh += 32) {
            const int khc = kh >> 3;      // K-half offset in 16B chunks
            // A 16x32 bf16 fragments: lane<16 -> K{0..7,16..23},
            //                         lane>=16 -> K{8..15,24..31}
            FragBf a0, a1, b0, b1;
            {
                int r0 = rbase + rlo;
                a0.u[0] = As4[r0 * 8 + khc + sel];
                a0.u[1] = As4[r0 * 8 + khc + 2 + sel];
                int r1 = r0 + 16;
                a1.u[0] = As4[r1 * 8 + khc + sel];
                a1.u[1] = As4[r1 * 8 + khc + 2 + sel];
            }
            // B 32x16 bf16: lanes 0-15 hold K{0..7,16..23}, lanes 16-31 K{8..15,24..31}
            {
                int c0 = cbase + rlo;
                b0.u[0] = Bs4[c0 * 8 + khc + sel];
                b0.u[1] = Bs4[c0 * 8 + khc + 2 + sel];
                int c1 = c0 + 16;
                b1.u[0] = Bs4[c1 * 8 + khc + sel];
                b1.u[1] = Bs4[c1 * 8 + khc + 2 + sel];
            }

            c00 = __builtin_amdgcn_wmma_f32_16x16x32_bf16(false, a0.v, false, b0.v,
                                                          (short)0, c00, false, false);
            c01 = __builtin_amdgcn_wmma_f32_16x16x32_bf16(false, a0.v, false, b1.v,
                                                          (short)0, c01, false, false);
            c10 = __builtin_amdgcn_wmma_f32_16x16x32_bf16(false, a1.v, false, b0.v,
                                                          (short)0, c10, false, false);
            c11 = __builtin_amdgcn_wmma_f32_16x16x32_bf16(false, a1.v, false, b1.v,
                                                          (short)0, c11, false, false);
        }

        if (it + 1 < NIT) {
            wait_async0();        // next buffer fully landed in LDS
            __syncthreads();      // (waits DScnt first -> safe to overwrite buf)
        }
    }

    // ---- epilogue: C layout (wave32): VGPR j -> M = j + 8*(lane>=16),
    //                N = lane & 15 ----
    const int rowoff = sel * 8;
    const int coln   = rlo;
    store_ctile(out, bias, c00, bm + rbase +  0 + rowoff, bn + cbase +  0 + coln);
    store_ctile(out, bias, c01, bm + rbase +  0 + rowoff, bn + cbase + 16 + coln);
    store_ctile(out, bias, c10, bm + rbase + 16 + rowoff, bn + cbase +  0 + coln);
    store_ctile(out, bias, c11, bm + rbase + 16 + rowoff, bn + cbase + 16 + coln);
}

// ------------------------------- launcher ----------------------------------
extern "C" void kernel_launch(void* const* d_in, const int* in_sizes, int n_in,
                              void* d_out, int out_size, void* d_ws, size_t ws_size,
                              hipStream_t stream) {
    const float* x     = (const float*)d_in[0];   // [1024,256]
    const float* sk    = (const float*)d_in[1];   // [256,8,256]
    const float* scale = (const float*)d_in[2];   // [256,256]
    const float* bias  = (const float*)d_in[3];   // [256]
    float* out         = (float*)d_out;           // [1024,256]

    __hip_bfloat16* A  = (__hip_bfloat16*)d_ws;                       // 4.50 MB
    __hip_bfloat16* Wt = (__hip_bfloat16*)((char*)d_ws +
                          (size_t)BATCH * KAUG * sizeof(__hip_bfloat16)); // +1.13 MB

    kan_prep_act<<<(BATCH * IN_DIM) / 256, 256, 0, stream>>>(x, A);
    kan_prep_w  <<<(UNITS * KAUG) / 256, 256, 0, stream>>>(sk, scale, Wt);

    dim3 grid(BATCH / BM, UNITS / BN);            // 8 x 4
    kan_gemm<<<grid, 256, 0, stream>>>(A, Wt, bias, out);
}
